// TransRenderer_82471962018592
// MI455X (gfx1250) — matrix operand order repaired and checked by reading, not compiled, not run on hardware
//
#include <hip/hip_runtime.h>
#include <hip/hip_bf16.h>
#include <math.h>

// Problem constants (fixed by reference)
#define BB   4
#define NN   4096
#define DD   128
#define BN   (BB * NN)      // 16384
#define KNB  16             // neighbors
#define HPOS 64
#define HMID 256
#define EPSN 1e-5f

typedef __attribute__((ext_vector_type(16))) _Float16 v16h;
typedef __attribute__((ext_vector_type(8)))  float    v8f;

// ---------------------------------------------------------------------------
// order-preserving key for float atomic max on unsigned
__device__ __forceinline__ unsigned orderKey(float f) {
    unsigned u = __float_as_uint(f);
    return (u & 0x80000000u) ? ~u : (u | 0x80000000u);
}
__device__ __forceinline__ float invOrderKey(unsigned u) {
    return (u & 0x80000000u) ? __uint_as_float(u ^ 0x80000000u)
                             : __uint_as_float(~u);
}

// ---------------------------------------------------------------------------
// Kernel A: pos embedding MLP  pos[BN,2] -> pemb[BN,128]
__global__ void posemb_kernel(const float* __restrict__ pos,
                              const float* __restrict__ w1,   // [64,2]
                              const float* __restrict__ b1,   // [64]
                              const float* __restrict__ w2,   // [128,64]
                              const float* __restrict__ b2,   // [128]
                              float* __restrict__ pemb) {
    __shared__ float pe[HPOS];
    const int row = blockIdx.x;
    const int t   = threadIdx.x;           // 128 threads
    const float px = pos[(size_t)row * 2 + 0];
    const float py = pos[(size_t)row * 2 + 1];
    if (t < HPOS) {
        float h = px * w1[t * 2 + 0] + py * w1[t * 2 + 1] + b1[t];
        pe[t] = fmaxf(h, 0.f);
    }
    __syncthreads();
    float s = b2[t];
    #pragma unroll 8
    for (int h = 0; h < HPOS; ++h) s += pe[h] * w2[t * HPOS + h];
    pemb[(size_t)row * DD + t] = s;
}

// ---------------------------------------------------------------------------
// WMMA GEMM: C[M,N] = epilogue(A[M,128] * W[N,128]^T + bias)
// fp32 in/out, f16 fragments, fp32 accumulate.
// One wave computes a 32(M) x 16(N) tile: two A fragments share one B
// fragment (weights are L2-resident; doubling M-tile doubles B reuse).
// K=128 is a compile-time constant -> the 4-step K loop fully unrolls to
// 8 static v_wmma_f32_16x16x32_f16 per instantiation.
// block = 256 threads = 8 waves over 8 consecutive 16-wide column tiles.
// grid = (M/32, N/128). Exact cover: EXEC stays all-ones around WMMA.
template<int NCOLS, bool RELU, bool BIAS>
__global__ void wmma_gemm_nt(const float* __restrict__ A,
                             const float* __restrict__ W,
                             const float* __restrict__ bias,
                             float* __restrict__ C,
                             int M) {
    constexpr int KDIM = 128;
    const int lane = threadIdx.x & 31;
    const int wave = threadIdx.x >> 5;
    const int grp  = lane >> 4;     // 0: lanes 0-15, 1: lanes 16-31
    const int mr   = lane & 15;     // A row / B col within tile
    const int m0   = blockIdx.x * 32;
    const int n0   = (blockIdx.y * 8 + wave) * 16;
    (void)M;

    v8f acc0 = {};
    v8f acc1 = {};
    const float* arow0 = A + (size_t)(m0 + mr) * KDIM;
    const float* arow1 = A + (size_t)(m0 + 16 + mr) * KDIM;
    const float* wrow  = W + (size_t)(n0 + mr) * KDIM;

    #pragma unroll
    for (int kk = 0; kk < KDIM; kk += 32) {
        // A fragment (16-bit A 16x32), contiguous k order shared with B:
        //   lane group g holds K = kk+16g+{0..15}
        v16h a0, a1;
        {
            const float* ap0 = arow0 + kk + grp * 16;
            const float* ap1 = arow1 + kk + grp * 16;
            #pragma unroll
            for (int j = 0; j < 8; ++j) {
                a0[j]     = (_Float16)ap0[j];
                a0[8 + j] = (_Float16)ap0[8 + j];
                a1[j]     = (_Float16)ap1[j];
                a1[8 + j] = (_Float16)ap1[8 + j];
            }
            __builtin_prefetch(ap0 + 32, 0, 1);   // global_prefetch next K chunk
        }
        // B fragment (B = W^T, 32x16): lane col = mr,
        //   group g holds K = kk + 16g + {0..15} (contiguous in W's row)
        v16h b;
        {
            const float* wp = wrow + kk + grp * 16;
            #pragma unroll
            for (int j = 0; j < 16; ++j) b[j] = (_Float16)wp[j];
        }

        acc0 = __builtin_amdgcn_wmma_f32_16x16x32_f16(
            false, a0, false, b, (short)0, acc0, false, false);
        acc1 = __builtin_amdgcn_wmma_f32_16x16x32_f16(
            false, a1, false, b, (short)0, acc1, false, false);
    }

    const int col = n0 + mr;
    float bv = 0.f;
    if (BIAS) bv = bias[col];
    #pragma unroll
    for (int r = 0; r < 8; ++r) {
        float v0 = acc0[r] + bv;
        float v1 = acc1[r] + bv;
        if (RELU) { v0 = fmaxf(v0, 0.f); v1 = fmaxf(v1, 0.f); }
        // C layout: VGPR r -> row r + 8*grp, col = lane%16
        C[(size_t)(m0 + r + grp * 8) * NCOLS + col]      = v0;
        C[(size_t)(m0 + 16 + r + grp * 8) * NCOLS + col] = v1;
    }
}

// ---------------------------------------------------------------------------
// Kernel C: exact top-16 Chebyshev kNN. One thread per query, positions of
// the whole batch staged in LDS (4096*2*4 = 32KB of the 320KB WGP pool).
// Register-resident sorted insertion (fully unrolled -> no scratch).
__global__ void knn_kernel(const float* __restrict__ pos,
                           int* __restrict__ idxb) {
    __shared__ float sx[NN];
    __shared__ float sy[NN];
    const int b = blockIdx.y;
    const float* pb = pos + (size_t)b * NN * 2;
    for (int j = threadIdx.x; j < NN; j += blockDim.x) {
        sx[j] = pb[2 * j + 0];
        sy[j] = pb[2 * j + 1];
    }
    __syncthreads();

    const int i  = blockIdx.x * blockDim.x + threadIdx.x;
    const float qx = sx[i], qy = sy[i];

    float best[KNB];
    int   bi[KNB];
    #pragma unroll
    for (int k = 0; k < KNB; ++k) { best[k] = 3.0e38f; bi[k] = 0; }

    for (int j = 0; j < NN; ++j) {
        const float d = fmaxf(fabsf(sx[j] - qx), fabsf(sy[j] - qy));
        if (d < best[KNB - 1]) {
            #pragma unroll
            for (int p = KNB - 1; p > 0; --p) {
                const bool sh = best[p - 1] > d;     // shift vs insert
                const float nb = sh ? best[p - 1] : d;
                const int   ni = sh ? bi[p - 1]   : j;
                if (best[p] > d) { best[p] = nb; bi[p] = ni; }
            }
            if (best[0] > d) { best[0] = d; bi[0] = j; }
        }
    }
    int* o = idxb + (size_t)(b * NN + i) * KNB;
    #pragma unroll
    for (int k = 0; k < KNB; ++k) o[k] = bi[k];
}

// ---------------------------------------------------------------------------
// Kernel D: sim[b,i,k] = q_i . k_{idx}, plus global max over i per (b,k).
// One wave (32 lanes) per query row; lane owns 4 of the 128 dims.
__global__ void sim_kernel(const float* __restrict__ qkv,   // [BN,384]
                           const int* __restrict__ idxb,
                           float* __restrict__ simb,
                           unsigned* __restrict__ maxu) {
    const int lane = threadIdx.x & 31;
    const int wv   = threadIdx.x >> 5;
    const int row  = blockIdx.x * 8 + wv;
    const int b    = row >> 12;                       // N = 4096
    const float4 qv = *(const float4*)(qkv + (size_t)row * 384 + lane * 4);
    const int* nb = idxb + (size_t)row * KNB;
    #pragma unroll 4
    for (int k = 0; k < KNB; ++k) {
        const size_t g = (size_t)(b * NN + nb[k]);
        const float4 kv = *(const float4*)(qkv + g * 384 + 128 + lane * 4);
        float p = qv.x * kv.x + qv.y * kv.y + qv.z * kv.z + qv.w * kv.w;
        #pragma unroll
        for (int o = 16; o >= 1; o >>= 1) p += __shfl_xor(p, o, 32);
        if (lane == 0) {
            simb[(size_t)row * KNB + k] = p;
            atomicMax(&maxu[b * KNB + k], orderKey(p));
        }
    }
}

// ---------------------------------------------------------------------------
// Kernel E: e = exp(sim - max[b,k]); accumulate S[b,k]. Overwrites simb.
__global__ void expsum_kernel(float* __restrict__ simb,
                              const unsigned* __restrict__ maxu,
                              float* __restrict__ S) {
    const int i = blockIdx.x * blockDim.x + threadIdx.x;   // BN*16 elems
    const int k = i & (KNB - 1);
    const int b = i >> 16;                                 // 4096*16 per batch
    const float m = invOrderKey(maxu[b * KNB + k]);
    const float e = __expf(simb[i] - m);
    simb[i] = e;
    atomicAdd(&S[b * KNB + k], e);
}

// ---------------------------------------------------------------------------
// Kernel F: agg = sum_k attn * (v_g + pe_g) + x, then InstanceNorm over D.
// One wave per row; lane owns 4 dims; wave32 shfl_xor reductions.
__global__ void agg_norm_kernel(const float* __restrict__ x,
                                const float* __restrict__ qkv,
                                const float* __restrict__ pemb,
                                const int* __restrict__ idxb,
                                const float* __restrict__ simb, // holds e
                                const float* __restrict__ S,
                                float* __restrict__ agg) {
    const int lane = threadIdx.x & 31;
    const int wv   = threadIdx.x >> 5;
    const int row  = blockIdx.x * 8 + wv;
    const int b    = row >> 12;
    float4 acc = *(const float4*)(x + (size_t)row * DD + lane * 4);
    const int* nb = idxb + (size_t)row * KNB;
    const float* eb = simb + (size_t)row * KNB;
    #pragma unroll 4
    for (int k = 0; k < KNB; ++k) {
        const float w = eb[k] / S[b * KNB + k];
        const size_t g = (size_t)(b * NN + nb[k]);
        const float4 vv = *(const float4*)(qkv  + g * 384 + 256 + lane * 4);
        const float4 pe = *(const float4*)(pemb + g * DD + lane * 4);
        acc.x += w * (vv.x + pe.x);
        acc.y += w * (vv.y + pe.y);
        acc.z += w * (vv.z + pe.z);
        acc.w += w * (vv.w + pe.w);
    }
    float s1 = acc.x + acc.y + acc.z + acc.w;
    float s2 = acc.x * acc.x + acc.y * acc.y + acc.z * acc.z + acc.w * acc.w;
    #pragma unroll
    for (int o = 16; o >= 1; o >>= 1) {
        s1 += __shfl_xor(s1, o, 32);
        s2 += __shfl_xor(s2, o, 32);
    }
    const float mean = s1 * (1.f / DD);
    const float var  = s2 * (1.f / DD) - mean * mean;
    const float inv  = rsqrtf(var + EPSN);
    float4 o4;
    o4.x = (acc.x - mean) * inv;
    o4.y = (acc.y - mean) * inv;
    o4.z = (acc.z - mean) * inv;
    o4.w = (acc.w - mean) * inv;
    *(float4*)(agg + (size_t)row * DD + lane * 4) = o4;
}

// ---------------------------------------------------------------------------
// Kernel H: out[row, 0..2] = sigmoid(h[row,:] @ w2^T + b2). One wave per row.
__global__ void out_kernel(const float* __restrict__ h,     // [BN,256]
                           const float* __restrict__ w2,    // [3,256]
                           const float* __restrict__ b2,    // [3]
                           float* __restrict__ out) {
    const int lane = threadIdx.x & 31;
    const int wv   = threadIdx.x >> 5;
    const int row  = blockIdx.x * 8 + wv;
    const float* hr = h + (size_t)row * HMID;
    float a0 = 0.f, a1 = 0.f, a2 = 0.f;
    #pragma unroll
    for (int c = lane; c < HMID; c += 32) {
        const float hv = hr[c];
        a0 += hv * w2[c];
        a1 += hv * w2[HMID + c];
        a2 += hv * w2[2 * HMID + c];
    }
    #pragma unroll
    for (int o = 16; o >= 1; o >>= 1) {
        a0 += __shfl_xor(a0, o, 32);
        a1 += __shfl_xor(a1, o, 32);
        a2 += __shfl_xor(a2, o, 32);
    }
    if (lane == 0) {
        out[(size_t)row * 3 + 0] = 1.f / (1.f + __expf(-(a0 + b2[0])));
        out[(size_t)row * 3 + 1] = 1.f / (1.f + __expf(-(a1 + b2[1])));
        out[(size_t)row * 3 + 2] = 1.f / (1.f + __expf(-(a2 + b2[2])));
    }
}

// ---------------------------------------------------------------------------
extern "C" void kernel_launch(void* const* d_in, const int* in_sizes, int n_in,
                              void* d_out, int out_size, void* d_ws, size_t ws_size,
                              hipStream_t stream) {
    (void)in_sizes; (void)n_in; (void)out_size; (void)ws_size;
    const float* x      = (const float*)d_in[0];
    const float* pos    = (const float*)d_in[1];
    const float* W_qkv  = (const float*)d_in[2];
    const float* pw1    = (const float*)d_in[3];
    const float* pb1    = (const float*)d_in[4];
    const float* pw2    = (const float*)d_in[5];
    const float* pb2    = (const float*)d_in[6];
    const float* mw1    = (const float*)d_in[7];
    const float* mb1    = (const float*)d_in[8];
    const float* mw2    = (const float*)d_in[9];
    const float* mb2    = (const float*)d_in[10];
    float* out = (float*)d_out;

    // workspace layout (floats)
    float* ws    = (float*)d_ws;
    float* qkv   = ws;                               // BN*384
    float* pemb  = qkv  + (size_t)BN * 384;          // BN*128
    float* agg   = pemb + (size_t)BN * DD;           // BN*128
    float* hbuf  = agg  + (size_t)BN * DD;           // BN*256
    float* simb  = hbuf + (size_t)BN * HMID;         // BN*16
    int*   idxb  = (int*)(simb + (size_t)BN * KNB);  // BN*16
    unsigned* maxu = (unsigned*)(idxb + (size_t)BN * KNB); // 64
    float* S     = (float*)(maxu + BB * KNB);        // 64

    hipMemsetAsync(maxu, 0, (BB * KNB) * (sizeof(unsigned) + sizeof(float)), stream);

    // pos embedding MLP
    posemb_kernel<<<BN, 128, 0, stream>>>(pos, pw1, pb1, pw2, pb2, pemb);

    // qkv projection: [BN,384] = x[BN,128] @ W_qkv[384,128]^T  (WMMA)
    wmma_gemm_nt<384, false, false><<<dim3(BN / 32, 384 / 128), 256, 0, stream>>>(
        x, W_qkv, nullptr, qkv, BN);

    // kNN (Chebyshev top-16)
    knn_kernel<<<dim3(NN / 256, BB), 256, 0, stream>>>(pos, idxb);

    // sim + global max over query axis
    sim_kernel<<<BN / 8, 256, 0, stream>>>(qkv, idxb, simb, maxu);

    // exp + global sum
    expsum_kernel<<<(BN * KNB) / 256, 256, 0, stream>>>(simb, maxu, S);

    // aggregate + residual + InstanceNorm
    agg_norm_kernel<<<BN / 8, 256, 0, stream>>>(x, qkv, pemb, idxb, simb, S, agg);

    // FFN layer 1: [BN,256] = relu(agg @ mlp_w1^T + b1)  (WMMA)
    wmma_gemm_nt<256, true, true><<<dim3(BN / 32, 256 / 128), 256, 0, stream>>>(
        agg, mw1, mb1, hbuf, BN);

    // output projection + sigmoid
    out_kernel<<<BN / 8, 256, 0, stream>>>(hbuf, mw2, mb2, out);
}
